// PGNet_54709293416656
// MI455X (gfx1250) — compile-verified, matrix operand order checked by reading, not run on hardware
//
#include <hip/hip_runtime.h>

// ---------------------------------------------------------------- constants
#define T_DIM  4096
#define B_DIM  8
#define N_DIM  256
#define H_DIM  512
#define NT_DIM 5

typedef __attribute__((ext_vector_type(16))) _Float16 v16h;
typedef __attribute__((ext_vector_type(8)))  _Float16 v8h;
typedef __attribute__((ext_vector_type(8)))  float    v8f;

// ------------------------------------------------------------ fragment load
// 16-bit A-matrix 16x32 fragment layout (CDNA5 ISA §7.12.2):
//   lanes 0-15 : row M=lane,    K = {0..7} and {16..23}
//   lanes 16-31: row M=lane-16, K = {8..15} and {24..31}
// B fragment (32x16, K-major) is the same gather applied to B^T (NxK rows).
__device__ __forceinline__ v16h load_frag16(const _Float16* __restrict__ p0,
                                            long rowStride, int lane) {
  const int r  = lane & 15;
  const int kb = (lane & 16) ? 8 : 0;
  const _Float16* p = p0 + (long)r * rowStride + kb;
  v8h lo = *(const v8h*)(p);        // K kb..kb+7   (16B aligned)
  v8h hi = *(const v8h*)(p + 16);   // K kb+16..kb+23
  v16h f;
#pragma unroll
  for (int i = 0; i < 8; ++i) { f[i] = lo[i]; f[i + 8] = hi[i]; }
  return f;
}

// --------------------------------------------------- 64x64 blocked WMMA GEMM
// C[z, m, n] = alpha * sum_k A[z, m, k] * Bt[z, n, k]  (+ resid)
// One wave32 per 64x64 C macro-tile = 4x4 grid of 16x16 WMMA tiles.
// Per K-step: 4 A-frags + 4 B-frags feed 16 WMMAs (32 FLOP/byte).
// K multiple of 32; M,N multiples of 64.
template <int K, bool OUT_F16, bool RESID>
__global__ __launch_bounds__(32)
void wmma_gemm64_kernel(const _Float16* __restrict__ A,  long lda, long aBatch,
                        const _Float16* __restrict__ Bt, long ldb, long bBatch,
                        void* __restrict__ Cv, long ldcM, long ldcN, long cBatch,
                        const float* __restrict__ resid, float alpha) {
  const int  lane = threadIdx.x & 31;
  const long tn = blockIdx.x, tm = blockIdx.y, z = blockIdx.z;
  const _Float16* Ab = A  + aBatch * z + tm * 64 * lda;
  const _Float16* Bb = Bt + bBatch * z + tn * 64 * ldb;

  v8f c[4][4];
#pragma unroll
  for (int i = 0; i < 4; ++i)
#pragma unroll
    for (int j = 0; j < 4; ++j) c[i][j] = (v8f){0.f,0.f,0.f,0.f,0.f,0.f,0.f,0.f};

  for (int k0 = 0; k0 < K; k0 += 32) {
    v16h a[4], b[4];
#pragma unroll
    for (int i = 0; i < 4; ++i) {
      __builtin_prefetch(Ab + (long)i * 16 * lda + k0 + 64, 0, 1);
      a[i] = load_frag16(Ab + (long)i * 16 * lda + k0, lda, lane);
    }
#pragma unroll
    for (int j = 0; j < 4; ++j)
      b[j] = load_frag16(Bb + (long)j * 16 * ldb + k0, ldb, lane);
#pragma unroll
    for (int i = 0; i < 4; ++i)
#pragma unroll
      for (int j = 0; j < 4; ++j)
        c[i][j] = __builtin_amdgcn_wmma_f32_16x16x32_f16(false, a[i], false, b[j],
                                                         (short)0, c[i][j],
                                                         false, false);
  }

  // C/D layout: VGPR r -> row (i*16 + r + (lane>=16 ? 8:0)), col j*16+(lane&15)
  const int colL = lane & 15;
  const int rOff = (lane & 16) ? 8 : 0;
#pragma unroll
  for (int i = 0; i < 4; ++i)
#pragma unroll
    for (int j = 0; j < 4; ++j) {
      const long row0 = tm * 64 + i * 16 + rOff;
      const long col  = tn * 64 + j * 16 + colL;
#pragma unroll
      for (int r = 0; r < 8; ++r) {
        const long idx = cBatch * z + (row0 + r) * ldcM + col * ldcN;
        float v = c[i][j][r] * alpha;
        if (RESID) v += resid[idx];
        if (OUT_F16) ((_Float16*)Cv)[idx] = (_Float16)v;
        else         ((float*)Cv)[idx]    = v;
      }
    }
}

// ------------------------------------------------------------- prep kernels
__global__ __launch_bounds__(256)
void cvt_f32_to_f16(const float* __restrict__ in, _Float16* __restrict__ out, long n) {
  long i      = (long)blockIdx.x * blockDim.x + threadIdx.x;
  long stride = (long)gridDim.x * blockDim.x;
  for (; i < n; i += stride) out[i] = (_Float16)in[i];
}

// Wt[n*H + k] = (f16) W[k*H + n]   (B^T layout for fragment loads)
__global__ __launch_bounds__(256)
void transpose_f32_to_f16(const float* __restrict__ W, _Float16* __restrict__ Wt, int H) {
  int i = blockIdx.x * 256 + threadIdx.x;
  int n = i / H, k = i - n * H;
  Wt[i] = (_Float16)W[k * H + n];
}

// -------------------------------------------------- combine + masked softmax
// combined = logit + sigmoid(aw) * other_logit (both already /sqrt(H))
// writes combined coalesced [B,T,N] into nm scratch, attn (f16) [B,T,N].
__global__ __launch_bounds__(256)
void combine_softmax_kernel(const float* __restrict__ lg, const float* __restrict__ lo,
                            const int* __restrict__ mask, const float* __restrict__ aw,
                            float* __restrict__ nm, _Float16* __restrict__ attn) {
  const int t = blockIdx.x, b = blockIdx.y, n = threadIdx.x;
  const float sig  = 1.0f / (1.0f + __expf(-aw[0]));
  const long  base = ((long)b * T_DIM + t) * N_DIM;
  const float comb = lg[base + n] + sig * lo[base + n];
  nm[base + n] = comb;
  const bool on = mask[b * N_DIM + n] > 0;
  __shared__ float red[256];
  red[n] = on ? comb : -3.0e38f;
  __syncthreads();
  for (int s = 128; s > 0; s >>= 1) {
    if (n < s) red[n] = fmaxf(red[n], red[n + s]);
    __syncthreads();
  }
  const float mx = red[0];
  __syncthreads();
  const float e = on ? __expf(comb - mx) : 0.0f;
  red[n] = e;
  __syncthreads();
  for (int s = 128; s > 0; s >>= 1) {
    if (n < s) red[n] += red[n + s];
    __syncthreads();
  }
  attn[base + n] = (_Float16)(e / red[0]);
}

// ------------------------------------- [B,T,N] -> [B,N,T] tiled transpose
// 32x32 LDS tiles, coalesced loads and stores. Block = 256 threads (32x8).
__global__ __launch_bounds__(256)
void transpose_TN_kernel(const float* __restrict__ in, float* __restrict__ out) {
  __shared__ float tile[32][33];
  const int b  = blockIdx.z;
  const int t0 = blockIdx.x * 32, n0 = blockIdx.y * 32;
  const int tx = threadIdx.x & 31, ty = threadIdx.x >> 5;
#pragma unroll
  for (int i = ty; i < 32; i += 8)
    tile[i][tx] = in[((long)b * T_DIM + t0 + i) * N_DIM + n0 + tx];
  __syncthreads();
#pragma unroll
  for (int i = ty; i < 32; i += 8)
    out[((long)b * N_DIM + n0 + i) * T_DIM + t0 + tx] = tile[tx][i];
}

// --------------------------------------------------------------- task head
__global__ __launch_bounds__(256)
void task_head_kernel(const float* __restrict__ task, const float* __restrict__ tw,
                      const float* __restrict__ tb, float* __restrict__ out, int H) {
  const int kb = blockIdx.x;           // k*B + b
  const float* row = task + (long)kb * H;
  __shared__ float red[256];
  float s = 0.f;
  for (int i = threadIdx.x; i < H; i += 256) s += row[i] * tw[i];
  red[threadIdx.x] = s;
  __syncthreads();
  for (int st = 128; st > 0; st >>= 1) {
    if (threadIdx.x < st) red[threadIdx.x] += red[threadIdx.x + st];
    __syncthreads();
  }
  if (threadIdx.x == 0) out[kb] = red[0] + tb[0];
}

// ------------------------------------------------------------------- launch
extern "C" void kernel_launch(void* const* d_in, const int* in_sizes, int n_in,
                              void* d_out, int out_size, void* d_ws, size_t ws_size,
                              hipStream_t stream) {
  (void)in_sizes; (void)n_in; (void)out_size; (void)ws_size;
  const long T = T_DIM, B = B_DIM, N = N_DIM, H = H_DIM;
  const long TB = T * B, NB = N * B;

  const float* actions[2] = { (const float*)d_in[0], (const float*)d_in[4] };
  const float* frames[2]  = { (const float*)d_in[1], (const float*)d_in[5] };
  const float* tasks[2]   = { (const float*)d_in[2], (const float*)d_in[6] };
  const int*   masks[2]   = { (const int*)d_in[3],   (const int*)d_in[7]   };
  // weights: 8..11 = v_Wq,v_Wk,v_Wv,v_Wo ; 12..15 = t_*
  const float* aws[2] = { (const float*)d_in[16], (const float*)d_in[17] };
  const float* tws[2] = { (const float*)d_in[18], (const float*)d_in[20] };
  const float* tbs[2] = { (const float*)d_in[19], (const float*)d_in[21] };

  // ---- workspace layout (256B-aligned regions)
  char* ws = (char*)d_ws;
  size_t off = 0;
  auto take = [&](size_t bytes) -> char* {
    char* p = ws + off;
    off += (bytes + 255) & ~(size_t)255;
    return p;
  };
  _Float16* WT[8];
  for (int i = 0; i < 8; ++i) WT[i] = (_Float16*)take(H * H * 2);
  _Float16* frame_h[2]  = { (_Float16*)take(TB * H * 2), (_Float16*)take(TB * H * 2) };
  _Float16* action_h[2] = { (_Float16*)take(NB * H * 2), (_Float16*)take(NB * H * 2) };
  _Float16* Qh    = (_Float16*)take(TB * H * 2);
  _Float16* Kh    = (_Float16*)take(NB * H * 2);
  _Float16* Vth   = (_Float16*)take(B * H * N * 2);   // [B][H][N]
  _Float16* attnh = (_Float16*)take(B * T * N * 2);   // [B][T][N]
  _Float16* midh  = (_Float16*)take(TB * H * 2);      // [T][B][H]
  float* logit[2] = { (float*)take(B * T * N * 4), (float*)take(B * T * N * 4) };
  float* nm_s     = (float*)take(B * T * N * 4);      // combined, [B,T,N]

  // ---- d_out layout: v_out, t_out, v_nm, t_nm, v_ta, t_ta
  float* outf = (float*)d_out;
  float* o_out[2] = { outf, outf + TB * H };
  float* o_nm[2]  = { outf + 2 * TB * H, outf + 2 * TB * H + B * N * T };
  float* o_ta[2]  = { outf + 2 * TB * H + 2 * B * N * T,
                      outf + 2 * TB * H + 2 * B * N * T + NT_DIM * B };

  // ---- prep: f16 conversions + weight transposes
  for (int d = 0; d < 2; ++d) {
    cvt_f32_to_f16<<<2048, 256, 0, stream>>>(frames[d],  frame_h[d],  TB * H);
    cvt_f32_to_f16<<<256,  256, 0, stream>>>(actions[d], action_h[d], NB * H);
  }
  for (int i = 0; i < 8; ++i)
    transpose_f32_to_f16<<<(H * H) / 256, 256, 0, stream>>>((const float*)d_in[8 + i],
                                                            WT[i], (int)H);

  const float inv_sqrt_h = 0.04419417382415922f;  // 1/sqrt(512)

  // ---- logits for both domains (needed cross-domain before outputs)
  for (int d = 0; d < 2; ++d) {
    // Q = frame @ Wq  -> f16 [T,B,H]
    wmma_gemm64_kernel<H_DIM, true, false><<<dim3(H / 64, TB / 64, 1), 32, 0, stream>>>(
        frame_h[d], H, 0, WT[4 * d + 0], H, 0, Qh, H, 1, 0, nullptr, 1.0f);
    // K = action @ Wk -> f16 [N,B,H]
    wmma_gemm64_kernel<H_DIM, true, false><<<dim3(H / 64, NB / 64, 1), 32, 0, stream>>>(
        action_h[d], H, 0, WT[4 * d + 1], H, 0, Kh, H, 1, 0, nullptr, 1.0f);
    // logits[b,t,n] = Q[t,b,:].K[n,b,:] / sqrt(H)   (batched, strided rows)
    wmma_gemm64_kernel<H_DIM, false, false><<<dim3(N / 64, T / 64, B), 32, 0, stream>>>(
        Qh, B * H, H, Kh, B * H, H, logit[d], N, 1, T * N, nullptr, inv_sqrt_h);
  }

  // ---- per-domain: combine+softmax, transpose, V proj, attn@V, @Wo+residual
  for (int d = 0; d < 2; ++d) {
    combine_softmax_kernel<<<dim3(T, B), 256, 0, stream>>>(
        logit[d], logit[1 - d], masks[d], aws[d], nm_s, attnh);
    transpose_TN_kernel<<<dim3(T / 32, N / 32, B), 256, 0, stream>>>(nm_s, o_nm[d]);
    // Vt[b,f,n] = sum_k action[n,b,k] Wv[k,f]  (C written transposed)
    wmma_gemm64_kernel<H_DIM, true, false><<<dim3(H / 64, N / 64, B), 32, 0, stream>>>(
        action_h[d], B * H, H, WT[4 * d + 2], H, 0, Vth, 1, N, H * N, nullptr, 1.0f);
    // mid[t,b,f] = sum_n attn[b,t,n] Vt[b,f,n]
    wmma_gemm64_kernel<N_DIM, true, false><<<dim3(H / 64, T / 64, B), 32, 0, stream>>>(
        attnh, N, T * N, Vth, N, H * N, midh, B * H, 1, H, nullptr, 1.0f);
    // out[t,b,g] = sum_f mid[t,b,f] Wo[f,g] + frame[t,b,g]
    wmma_gemm64_kernel<H_DIM, false, true><<<dim3(H / 64, TB / 64, 1), 32, 0, stream>>>(
        midh, H, 0, WT[4 * d + 3], H, 0, o_out[d], H, 1, 0, frames[d], 1.0f);
  }

  // ---- task heads: [NT,B,1]
  for (int d = 0; d < 2; ++d)
    task_head_kernel<<<NT_DIM * B, 256, 0, stream>>>(tasks[d], tws[d], tbs[d],
                                                     o_ta[d], (int)H);
}
